// SelfModifyingTitansLite_33148557591245
// MI455X (gfx1250) — matrix operand order repaired and checked by reading, not compiled, not run on hardware
//
#include <hip/hip_runtime.h>
#include <math.h>

// Problem constants
#define BATCH 32
#define SEQ   8192
#define DIM   256
#define WIN   4

#define LDH   264          // padded LDS row stride in halves (264*2=528B, bank-shift 4)
#define NW    8            // waves per block
#define MROWS 128          // rows per workgroup (NW * 16)
#define PROWS 128          // weight panel rows (half a 256x256 matrix)

typedef __attribute__((ext_vector_type(16))) _Float16 v16h;
typedef __attribute__((ext_vector_type(8)))  _Float16 v8h;
typedef __attribute__((ext_vector_type(8)))  float    v8f;
typedef __attribute__((ext_vector_type(4)))  int      v4i;

typedef v4i __attribute__((address_space(1))) gv4i;   // global 16B chunk
typedef v4i __attribute__((address_space(3))) lv4i;   // LDS 16B chunk

#if __has_builtin(__builtin_amdgcn_global_load_async_to_lds_b128)
#define HAVE_ASYNC_LDS 1
#else
#define HAVE_ASYNC_LDS 0
#endif

__device__ __forceinline__ v16h cat8(v8h a, v8h b) {
  return __builtin_shufflevector(a, b, 0,1,2,3,4,5,6,7,8,9,10,11,12,13,14,15);
}

// A-matrix (16x32 f16) fragment: lane holds row (lane&15);
// halves 0..7 = K = ko+0..7, halves 8..15 = K = 16+ko+0..7, ko = (lane>>4)*8
__device__ __forceinline__ v16h load_fragA(const _Float16* base, int lane, int k) {
  const int r  = lane & 15;
  const int ko = (lane >> 4) << 3;
  const _Float16* p = base + r * LDH + k + ko;
  return cat8(*(const v8h*)p, *(const v8h*)(p + 16));
}

// B-matrix (32x16 f16) fragment for Y = A @ W^T read from row-major W:
// lane holds column n = n0 + (lane&15) (a row of W); halves 0..7 = K = ko+0..7, halves 8..15 = K = 16+ko+0..7, ko = (lane>>4)*8
__device__ __forceinline__ v16h load_fragB(const _Float16* base, int lane, int n0, int k) {
  const int r  = n0 + (lane & 15);
  const int ko = (lane >> 4) << 3;
  const _Float16* p = base + r * LDH + k + ko;
  return cat8(*(const v8h*)p, *(const v8h*)(p + 16));
}

// one 16x16 output tile: acc += A(16x256) x W_panel^T tile
__device__ __forceinline__ v8f mm_tile(const v16h af[8], const _Float16* w,
                                       int lane, int n0) {
  v8f acc = {};
  #pragma unroll
  for (int kk = 0; kk < 8; ++kk) {
    const v16h bf = load_fragB(w, lane, n0, kk * 32);
    acc = __builtin_amdgcn_wmma_f32_16x16x32_f16(false, af[kk], false, bf,
                                                 (short)0, acc, false, false);
  }
  return acc;
}

__device__ __forceinline__ float red16(float x) {
  x += __shfl_xor(x, 1);
  x += __shfl_xor(x, 2);
  x += __shfl_xor(x, 4);
  x += __shfl_xor(x, 8);
  return x;
}

__device__ __forceinline__ float gelu_exact(float x) {
  return 0.5f * x * (1.0f + erff(x * 0.70710678118654752440f));
}

// ---- async (ASYNCcnt) weight-panel staging: 128 rows x 256 halves -> padded LDS
// each wave issues exactly 16 async instructions per panel
__device__ __forceinline__ void issue_panel(const _Float16* __restrict__ g,
                                            _Float16* dst, int tid) {
  #pragma unroll
  for (int i = tid; i < PROWS * 32; i += 256) {   // 4096 16-byte chunks
    const int row = i >> 5;
    const int c   = (i & 31) << 3;
#if HAVE_ASYNC_LDS
    __builtin_amdgcn_global_load_async_to_lds_b128(
        (gv4i*)&g[row * 256 + c],
        (lv4i*)&dst[row * LDH + c],
        0, 0);
#else
    *(v8h*)&dst[row * LDH + c] = *(const v8h*)&g[row * 256 + c];
#endif
  }
}

__device__ __forceinline__ void wait_async16() {   // all but newest panel done
#if HAVE_ASYNC_LDS
#if __has_builtin(__builtin_amdgcn_s_wait_asynccnt)
  __builtin_amdgcn_s_wait_asynccnt(16);
#else
  asm volatile("s_wait_asynccnt 0x10" ::: "memory");
#endif
#endif
}
__device__ __forceinline__ void wait_async0() {    // everything done
#if HAVE_ASYNC_LDS
#if __has_builtin(__builtin_amdgcn_s_wait_asynccnt)
  __builtin_amdgcn_s_wait_asynccnt(0);
#else
  asm volatile("s_wait_asynccnt 0x0" ::: "memory");
#endif
#endif
}

// --- kernel 0: fp32 -> fp16 weight conversion into workspace ------------------
__global__ __launch_bounds__(256) void cvt_weights_f16(
    const float* __restrict__ qw, const float* __restrict__ w2,
    const float* __restrict__ w1, _Float16* __restrict__ ws) {
  const int i = blockIdx.x * 256 + threadIdx.x;   // 0 .. 65535
  ws[i]                 = (_Float16)qw[i];
  ws[DIM * DIM + i]     = (_Float16)w2[i];
  ws[2 * DIM * DIM + i] = (_Float16)w1[i];
}

// --- fused kernel: conv+residual -> GEMM1+L2norm -> GEMM2+gelu -> GEMM3+skip --
__global__ __launch_bounds__(256) void fused_titans(
    const float* __restrict__ x, const float* __restrict__ conv_w,
    const _Float16* __restrict__ ws, float* __restrict__ out) {
  __shared__ __align__(16) _Float16 Wp[2][PROWS * LDH];    // 2 x 66 KB weight panels
  __shared__ __align__(16) _Float16 act[NW][2][16 * LDH];  // per-wave {h|hid, q}
  __shared__ float scl[NW][16];

  const int tid  = threadIdx.x;
  const int wave = tid >> 5;
  const int lane = tid & 31;
  const int lrow = lane & 15;
  const int hi8  = (lane >> 4) << 3;

  const int b  = blockIdx.x >> 6;          // 32 batches
  const int s0 = (blockIdx.x & 63) << 7;   // 64 chunks of 128 rows

  // panels: p0,p1 = q_w lo/hi; p2,p3 = mem_w2 lo/hi; p4,p5 = mem_w1 lo/hi
  issue_panel(ws,               Wp[0], tid);                 // p0
  issue_panel(ws + PROWS * DIM, Wp[1], tid);                 // p1

  // depthwise causal conv + residual; thread tid owns column d = tid
  {
    const float* xb = x + ((size_t)b * SEQ + s0) * DIM + tid;
    const float c0 = conv_w[tid * WIN + 0];
    const float c1 = conv_w[tid * WIN + 1];
    const float c2 = conv_w[tid * WIN + 2];
    const float c3 = conv_w[tid * WIN + 3];
    float xm3 = 0.f, xm2 = 0.f, xm1 = 0.f;
    if (s0 != 0) {
      xm3 = xb[-3 * DIM]; xm2 = xb[-2 * DIM]; xm1 = xb[-1 * DIM];
    }
    #pragma unroll 4
    for (int r = 0; r < MROWS; ++r) {
      const float xv = xb[(size_t)r * DIM];
      const float h  = xv + c0 * xm3 + c1 * xm2 + c2 * xm1 + c3 * xv;
      act[r >> 4][0][(r & 15) * LDH + tid] = (_Float16)h;
      xm3 = xm2; xm2 = xm1; xm1 = xv;
    }
  }
  wait_async16();            // p0 landed (this wave); barrier publishes all waves
  __syncthreads();

  _Float16* actH = &act[wave][0][0];   // conv output, later reused for gelu hidden
  _Float16* actQ = &act[wave][1][0];   // (unnormalized) q tile

  v16h af[8];

  // ================= GEMM1: q = h @ q_w^T, track row sum-of-squares ==========
  #pragma unroll
  for (int kk = 0; kk < 8; ++kk) af[kk] = load_fragA(actH, lane, kk * 32);

  float rowsq[8];
  #pragma unroll
  for (int v = 0; v < 8; ++v) rowsq[v] = 0.f;

  for (int nt = 0; nt < 8; ++nt) {             // cols 0..127 from panel p0
    const v8f acc = mm_tile(af, Wp[0], lane, nt * 16);
    #pragma unroll
    for (int v = 0; v < 8; ++v) {
      const float q = acc[v];
      rowsq[v] += q * q;
      actQ[(v + hi8) * LDH + nt * 16 + lrow] = (_Float16)q;
    }
  }
  __syncthreads();                              // all waves done with Wp[0]
  issue_panel(ws + DIM * DIM, Wp[0], tid);      // p2 = mem_w2 lo
  wait_async16();                               // p1 landed
  __syncthreads();

  for (int nt = 8; nt < 16; ++nt) {            // cols 128..255 from panel p1
    const v8f acc = mm_tile(af, Wp[1], lane, (nt - 8) * 16);
    #pragma unroll
    for (int v = 0; v < 8; ++v) {
      const float q = acc[v];
      rowsq[v] += q * q;
      actQ[(v + hi8) * LDH + nt * 16 + lrow] = (_Float16)q;
    }
  }
  #pragma unroll
  for (int v = 0; v < 8; ++v) {
    const float ss = red16(rowsq[v]);
    const float s  = 1.0f / fmaxf(sqrtf(ss), 1e-6f);
    if (lrow == 0) scl[wave][v + hi8] = s;
  }
  __syncthreads();                              // all waves done with Wp[1]
  issue_panel(ws + DIM * DIM + PROWS * DIM, Wp[1], tid);   // p3 = mem_w2 hi
  wait_async16();                               // p2 landed
  __syncthreads();

  // ================= GEMM2: hid = gelu(qn @ mem_w2^T) ========================
  {
    const _Float16 sch = (_Float16)scl[wave][lrow];   // A-frag lane = full row
    #pragma unroll
    for (int kk = 0; kk < 8; ++kk) {
      v16h f = load_fragA(actQ, lane, kk * 32);
      #pragma unroll
      for (int i = 0; i < 16; ++i) f[i] *= sch;       // fold L2-normalize into A
      af[kk] = f;
    }
  }
  for (int nt = 0; nt < 8; ++nt) {             // panel p2
    const v8f acc = mm_tile(af, Wp[0], lane, nt * 16);
    #pragma unroll
    for (int v = 0; v < 8; ++v)
      actH[(v + hi8) * LDH + nt * 16 + lrow] = (_Float16)gelu_exact(acc[v]);
  }
  __syncthreads();
  issue_panel(ws + 2 * DIM * DIM, Wp[0], tid);  // p4 = mem_w1 lo
  wait_async16();                               // p3 landed
  __syncthreads();

  for (int nt = 8; nt < 16; ++nt) {            // panel p3
    const v8f acc = mm_tile(af, Wp[1], lane, (nt - 8) * 16);
    #pragma unroll
    for (int v = 0; v < 8; ++v)
      actH[(v + hi8) * LDH + nt * 16 + lrow] = (_Float16)gelu_exact(acc[v]);
  }
  __syncthreads();
  issue_panel(ws + 2 * DIM * DIM + PROWS * DIM, Wp[1], tid); // p5 = mem_w1 hi
  wait_async16();                               // p4 landed
  __syncthreads();

  // ================= GEMM3: out = qn + hid @ mem_w1^T ========================
  #pragma unroll
  for (int kk = 0; kk < 8; ++kk) af[kk] = load_fragA(actH, lane, kk * 32);

  const size_t rowbase = (size_t)b * SEQ + s0 + wave * 16;
  for (int nt = 0; nt < 8; ++nt) {             // panel p4
    const v8f acc = mm_tile(af, Wp[0], lane, nt * 16);
    #pragma unroll
    for (int v = 0; v < 8; ++v) {
      const int m   = v + hi8;
      const int col = nt * 16 + lrow;
      const float qn = (float)actQ[m * LDH + col] * scl[wave][m];
      out[(rowbase + m) * DIM + col] = qn + acc[v];
    }
  }
  wait_async0();                                // p5 landed
  __syncthreads();

  for (int nt = 8; nt < 16; ++nt) {            // panel p5
    const v8f acc = mm_tile(af, Wp[1], lane, (nt - 8) * 16);
    #pragma unroll
    for (int v = 0; v < 8; ++v) {
      const int m   = v + hi8;
      const int col = nt * 16 + lrow;
      const float qn = (float)actQ[m * LDH + col] * scl[wave][m];
      out[(rowbase + m) * DIM + col] = qn + acc[v];
    }
  }
}

extern "C" void kernel_launch(void* const* d_in, const int* in_sizes, int n_in,
                              void* d_out, int out_size, void* d_ws, size_t ws_size,
                              hipStream_t stream) {
  const float* x      = (const float*)d_in[0];
  const float* conv_w = (const float*)d_in[1];
  const float* q_w    = (const float*)d_in[2];
  const float* mem_w2 = (const float*)d_in[3];
  const float* mem_w1 = (const float*)d_in[4];
  float*       out    = (float*)d_out;
  _Float16*    wsf16  = (_Float16*)d_ws;   // 3 * 256*256 halves = 384 KB

  cvt_weights_f16<<<DIM * DIM / 256, 256, 0, stream>>>(q_w, mem_w2, mem_w1, wsf16);

  const int nblocks = BATCH * (SEQ / MROWS);   // 32 * 64 = 2048
  fused_titans<<<nblocks, 256, 0, stream>>>(x, conv_w, wsf16, out);
}
